// _IPEXLlamaAttention_90099823935703
// MI455X (gfx1250) — matrix operand order repaired and checked by reading, not compiled, not run on hardware
//
#include <hip/hip_runtime.h>

#define BB 2
#define SS 2048
#define HH 4096
#define NH 32
#define NKV 8
#define HD 128
#define GROUPS 4
#define NCAT 6144            // 4096 (Q) + 1024 (K) + 1024 (V)
#define ROWS (BB*SS)         // 4096

typedef __attribute__((ext_vector_type(16))) __bf16 v16bf;
typedef __attribute__((ext_vector_type(8)))  float  v8f;
typedef __attribute__((ext_vector_type(4)))  unsigned u32x4;
typedef __attribute__((ext_vector_type(8)))  int      i32x8;
typedef __attribute__((ext_vector_type(4)))  int      i32x4;

union AF { v16bf v; unsigned u[8]; };

#if __has_builtin(__builtin_amdgcn_tensor_load_to_lds) && __has_builtin(__builtin_amdgcn_s_wait_tensorcnt)
#define HAVE_TDM 1
#endif

// ---------------- conversion kernels ----------------
__global__ void cvt_x(const float* __restrict__ x, __bf16* __restrict__ xb) {
  int i = blockIdx.x * 256 + threadIdx.x;          // ROWS*HH total, exact
  xb[i] = (__bf16)x[i];
}

__global__ void cvt_w(const float* __restrict__ wq, const float* __restrict__ wk,
                      const float* __restrict__ wv, __bf16* __restrict__ wc) {
  int i = blockIdx.x * 256 + threadIdx.x;          // HH*NCAT total, exact
  int r = i / NCAT, c = i - r * NCAT;
  float v;
  if (c < 4096)       v = wq[(size_t)r * 4096 + c];
  else if (c < 5120)  v = wk[(size_t)r * 1024 + (c - 4096)];
  else                v = wv[(size_t)r * 1024 + (c - 5120)];
  wc[i] = (__bf16)v;
}

// ---------------- fused QKV GEMM (bf16 WMMA, f32 accum) ----------------
// 256 threads = 8 wave32. Tile: M=128 (16 rows/wave), N=128 (8 n-tiles/wave), K-step 32.
__global__ __launch_bounds__(256) void qkv_gemm(const __bf16* __restrict__ X,
                                                const __bf16* __restrict__ W,
                                                __bf16* __restrict__ O) {
  __shared__ __bf16 Xs[128][32];    // [m][k] row-major
  __shared__ __bf16 Ws[128][32];    // [n][k]  (W^T: contiguous K pairs for B-fragments)
  const int t = threadIdx.x;
  const int lane = t & 31, w = t >> 5;
  const int l15 = lane & 15, hi = (lane >> 4) & 1;
  const int m0 = blockIdx.x * 128;
  const int n0 = blockIdx.y * 128;

  v8f acc[8] = {};
  for (int k0 = 0; k0 < HH; k0 += 32) {
    {   // stage X tile: thread loads 16 contiguous bf16 (two 16B loads)
      int row = t >> 1, half = t & 1;
      const uint4* src = (const uint4*)(X + (size_t)(m0 + row) * HH + k0 + half * 16);
      uint4 d0 = src[0], d1 = src[1];
      *(uint4*)&Xs[row][half * 16]     = d0;
      *(uint4*)&Xs[row][half * 16 + 8] = d1;
    }
    {   // stage W tile transposed: coalesced read of 16 n's, scatter to [n][k]
      int k = t >> 3, g = t & 7;
      const uint4* src = (const uint4*)(W + (size_t)(k0 + k) * NCAT + n0 + g * 16);
      uint4 d0 = src[0], d1 = src[1];
      const __bf16* e0 = (const __bf16*)&d0;
      const __bf16* e1 = (const __bf16*)&d1;
      #pragma unroll
      for (int i = 0; i < 8; ++i) Ws[g * 16 + i][k]     = e0[i];
      #pragma unroll
      for (int i = 0; i < 8; ++i) Ws[g * 16 + 8 + i][k] = e1[i];
    }
    if (k0 + 32 < HH) {   // prefetch next tiles (global_prefetch_b8)
      __builtin_prefetch(X + (size_t)(m0 + (t >> 1)) * HH + (k0 + 32) + (t & 1) * 16, 0, 0);
      __builtin_prefetch(W + (size_t)(k0 + 32 + (t >> 3)) * NCAT + n0 + (t & 7) * 16, 0, 0);
    }
    __syncthreads();

    AF a;   // A fragment: 16x32 bf16, lane-half K-pair layout
    #pragma unroll
    for (int j = 0; j < 8; ++j) {
      int klo = ((j & 3) << 1) + ((j >> 2) << 4) + (hi << 3);
      a.u[j] = *(const unsigned*)&Xs[w * 16 + l15][klo];
    }
    #pragma unroll
    for (int nt = 0; nt < 8; ++nt) {
      AF bf;  // B fragment: 32x16, K = 2(j&3) + 16(j>>2) (+8 for hi half)
      #pragma unroll
      for (int j = 0; j < 8; ++j) {
        int klo = ((j & 3) << 1) + ((j >> 2) << 4) + (hi << 3);
        bf.u[j] = *(const unsigned*)&Ws[nt * 16 + l15][klo];
      }
      acc[nt] = __builtin_amdgcn_wmma_f32_16x16x32_bf16(
          false, a.v, false, bf.v, (short)0, acc[nt], false, false);
    }
    __syncthreads();
  }
  // epilogue: D layout row = r + 8*hi, col = lane&15
  #pragma unroll
  for (int nt = 0; nt < 8; ++nt)
    #pragma unroll
    for (int r = 0; r < 8; ++r) {
      int m = m0 + w * 16 + r + hi * 8;
      int n = n0 + nt * 16 + l15;
      O[(size_t)m * NCAT + n] = (__bf16)acc[nt][r];
    }
}

// ---------------- RoPE + pack to [B,h,S,hd] bf16 ----------------
__global__ void rope_pack(const __bf16* __restrict__ T, __bf16* __restrict__ Qb,
                          __bf16* __restrict__ Kb, __bf16* __restrict__ Vb) {
  int tid = blockIdx.x * 256 + threadIdx.x;        // ROWS*48*64 total, exact
  int j   = tid & 63;
  int hh  = (tid >> 6) % 48;
  int row = tid / (64 * 48);
  int b = row / SS, s = row - b * SS;
  const __bf16* src = T + (size_t)row * NCAT;
  float inv = __powf(10000.0f, -(float)(2 * j) / 128.0f);
  float ang = (float)s * inv;
  float c = __cosf(ang), sn = __sinf(ang);
  if (hh < NH) {                 // Q head: RoPE + fold in 1/sqrt(hd)
    float a  = (float)src[hh * HD + j];
    float a2 = (float)src[hh * HD + j + 64];
    const float scl = 0.08838834764831845f;        // 1/sqrt(128)
    __bf16* dst = Qb + ((size_t)(b * NH + hh) * SS + s) * HD;
    dst[j]      = (__bf16)((a * c - a2 * sn) * scl);
    dst[j + 64] = (__bf16)((a2 * c + a * sn) * scl);
  } else if (hh < NH + NKV) {    // K head: RoPE
    int h = hh - NH;
    float a  = (float)src[4096 + h * HD + j];
    float a2 = (float)src[4096 + h * HD + j + 64];
    __bf16* dst = Kb + ((size_t)(b * NKV + h) * SS + s) * HD;
    dst[j]      = (__bf16)(a * c - a2 * sn);
    dst[j + 64] = (__bf16)(a2 * c + a * sn);
  } else {                       // V head: copy
    int h = hh - NH - NKV;
    __bf16* dst = Vb + ((size_t)(b * NKV + h) * SS + s) * HD;
    dst[j]      = src[5120 + h * HD + j];
    dst[j + 64] = src[5120 + h * HD + j + 64];
  }
}

// ---------------- causal flash attention (bf16 WMMA, f32 softmax) ----------------
// 256 threads = 8 wave32; 128 query rows per WG (16/wave); 32 keys per iteration.
// K tile is staged by the Tensor Data Mover (TDM) when available; V is staged
// transposed manually (TDM copies rows linearly, PV B-fragments need [hd][key]).
__global__ __launch_bounds__(256) void attn(const __bf16* __restrict__ Qb,
                                            const __bf16* __restrict__ Kb,
                                            const __bf16* __restrict__ Vb,
                                            float* __restrict__ Out) {
  __shared__ __bf16 Kt[32][HD];       // [key][hd]
  __shared__ __bf16 Vt[HD][32];       // [hd][key] (transposed for PV B-fragments)
  __shared__ __bf16 Pb[8][16][32];    // per-wave P scratch [m][k]
  const int t = threadIdx.x, lane = t & 31, w = t >> 5;
  const int l15 = lane & 15, hi = (lane >> 4) & 1;
  const int q0 = blockIdx.x * 128;
  const int h = blockIdx.y, b = blockIdx.z;
  const int kvh = h / GROUPS;
  const __bf16* Qp = Qb + ((size_t)(b * NH + h) * SS + q0 + w * 16) * HD;
  const __bf16* Kp = Kb + ((size_t)(b * NKV + kvh) * SS) * HD;
  const __bf16* Vp = Vb + ((size_t)(b * NKV + kvh) * SS) * HD;

  AF qf[4];                           // Q row strip, kept in registers
  #pragma unroll
  for (int c = 0; c < 4; ++c)
    #pragma unroll
    for (int j = 0; j < 8; ++j) {
      int klo = ((j & 3) << 1) + ((j >> 2) << 4) + (hi << 3);
      qf[c].u[j] = *(const unsigned*)(Qp + (size_t)l15 * HD + c * 32 + klo);
    }

  v8f oacc[8] = {};
  float mrow[8], lrow[8];
  #pragma unroll
  for (int r = 0; r < 8; ++r) { mrow[r] = -3.0e38f; lrow[r] = 0.0f; }

#ifdef HAVE_TDM
  const unsigned ldsK = (unsigned)(size_t)(void*)&Kt[0][0];   // LDS byte offset
#endif

  const int kvend = q0 + 128;         // causal: keys beyond the tile are all masked
  for (int kv0 = 0; kv0 < kvend; kv0 += 32) {
#ifdef HAVE_TDM
    if (w == 0) {     // one wave drives the TDM: K tile 32 keys x 128 hd, bf16
      unsigned long long ga = (unsigned long long)(size_t)(Kp + (size_t)kv0 * HD);
      u32x4 g0;
      g0[0] = 1u;                                   // count=1, user mode, no gather
      g0[1] = ldsK;                                 // lds_addr [63:32]
      g0[2] = (unsigned)(ga & 0xffffffffu);         // global_addr[31:0]
      g0[3] = (unsigned)((ga >> 32) & 0x01ffffffu)  // global_addr[56:32]
              | 0x80000000u;                        // type=2 ("image") [127:126]
      i32x8 g1;
      g1[0] = 0x00010000;                           // wg_mask=0, data_size=2B
      g1[1] = (int)(128u << 16);                    // tensor_dim0 = 128 (hd)
      g1[2] = (int)(2048u << 16);                   // tensor_dim1 = 2048 (keys)
      g1[3] = (int)(128u << 16);                    // tile_dim0 = 128
      g1[4] = 32;                                   // tile_dim1 = 32, tile_dim2 = 0
      g1[5] = 128;                                  // tensor_dim0_stride = 128
      g1[6] = 0; g1[7] = 0;
      i32x4 gz = {0, 0, 0, 0};
#if defined(__clang_major__) && __clang_major__ >= 23
      i32x8 gz8 = {0, 0, 0, 0, 0, 0, 0, 0};
      __builtin_amdgcn_tensor_load_to_lds(g0, g1, gz, gz, gz8, 0);
#else
      __builtin_amdgcn_tensor_load_to_lds(g0, g1, gz, gz, 0);
#endif
    }
#endif
    {   // stage V transposed (and K manually if no TDM)
      int key = t >> 3, seg = t & 7;
#ifndef HAVE_TDM
      const uint4* ks = (const uint4*)(Kp + (size_t)(kv0 + key) * HD + seg * 16);
      uint4 k0v = ks[0], k1v = ks[1];
      *(uint4*)&Kt[key][seg * 16]     = k0v;
      *(uint4*)&Kt[key][seg * 16 + 8] = k1v;
#endif
      const uint4* vs = (const uint4*)(Vp + (size_t)(kv0 + key) * HD + seg * 16);
      uint4 v0 = vs[0], v1 = vs[1];
      const __bf16* e0 = (const __bf16*)&v0;
      const __bf16* e1 = (const __bf16*)&v1;
      #pragma unroll
      for (int i = 0; i < 8; ++i) Vt[seg * 16 + i][key]     = e0[i];
      #pragma unroll
      for (int i = 0; i < 8; ++i) Vt[seg * 16 + 8 + i][key] = e1[i];
      if (kv0 + 32 < kvend)   // prefetch next V chunk
        __builtin_prefetch(Vp + (size_t)(kv0 + 32 + key) * HD + seg * 16, 0, 0);
    }
#ifdef HAVE_TDM
    if (w == 0) __builtin_amdgcn_s_wait_tensorcnt(0);
#endif
    __syncthreads();

    // S = Q K^T over hd=128 (4 chained WMMAs per 16-key tile)
    v8f sacc[2] = {};
    #pragma unroll
    for (int nt = 0; nt < 2; ++nt)
      #pragma unroll
      for (int c = 0; c < 4; ++c) {
        AF bf;
        #pragma unroll
        for (int j = 0; j < 8; ++j) {
          int klo = ((j & 3) << 1) + ((j >> 2) << 4) + (hi << 3);
          bf.u[j] = *(const unsigned*)&Kt[nt * 16 + l15][c * 32 + klo];
        }
        sacc[nt] = __builtin_amdgcn_wmma_f32_16x16x32_bf16(
            false, qf[c].v, false, bf.v, (short)0, sacc[nt], false, false);
      }

    // causal mask + online softmax (row = r + 8*hi, col = lane&15)
    #pragma unroll
    for (int r = 0; r < 8; ++r) {
      int qg = q0 + w * 16 + r + hi * 8;
      float s0 = sacc[0][r], s1 = sacc[1][r];
      if (kv0 + l15 > qg)      s0 = -3.0e38f;
      if (kv0 + 16 + l15 > qg) s1 = -3.0e38f;
      float mx = fmaxf(s0, s1);
      #pragma unroll
      for (int d = 1; d < 16; d <<= 1) mx = fmaxf(mx, __shfl_xor(mx, d, 32));
      float mnew  = fmaxf(mrow[r], mx);
      float alpha = __expf(mrow[r] - mnew);
      float p0 = __expf(s0 - mnew), p1 = __expf(s1 - mnew);
      float ps = p0 + p1;
      #pragma unroll
      for (int d = 1; d < 16; d <<= 1) ps += __shfl_xor(ps, d, 32);
      lrow[r] = lrow[r] * alpha + ps;
      mrow[r] = mnew;
      #pragma unroll
      for (int jj = 0; jj < 8; ++jj) oacc[jj][r] *= alpha;
      Pb[w][r + hi * 8][l15]      = (__bf16)p0;   // D-layout -> LDS
      Pb[w][r + hi * 8][16 + l15] = (__bf16)p1;
    }

    // O += P V  (A = P 16x32 re-read in A-layout; B from transposed V tile)
    AF pa;
    #pragma unroll
    for (int j = 0; j < 8; ++j) {
      int klo = ((j & 3) << 1) + ((j >> 2) << 4) + (hi << 3);
      pa.u[j] = *(const unsigned*)&Pb[w][l15][klo];
    }
    #pragma unroll
    for (int jj = 0; jj < 8; ++jj) {
      AF bf;
      #pragma unroll
      for (int j = 0; j < 8; ++j) {
        int klo = ((j & 3) << 1) + ((j >> 2) << 4) + (hi << 3);
        bf.u[j] = *(const unsigned*)&Vt[jj * 16 + l15][klo];
      }
      oacc[jj] = __builtin_amdgcn_wmma_f32_16x16x32_bf16(
          false, pa.v, false, bf.v, (short)0, oacc[jj], false, false);
    }
    __syncthreads();
  }

  // epilogue: out[b][s][h*128+hd] = acc / l
  #pragma unroll
  for (int r = 0; r < 8; ++r) {
    int qg = q0 + w * 16 + r + hi * 8;
    float inv = 1.0f / lrow[r];
    float* dst = Out + ((size_t)(b * SS + qg) * HH) + h * HD;
    #pragma unroll
    for (int jj = 0; jj < 8; ++jj)
      dst[jj * 16 + l15] = oacc[jj][r] * inv;
  }
}

// ---------------- launcher ----------------
extern "C" void kernel_launch(void* const* d_in, const int* in_sizes, int n_in,
                              void* d_out, int out_size, void* d_ws, size_t ws_size,
                              hipStream_t stream) {
  const float* X  = (const float*)d_in[0];
  // d_in[1] = position_ids (arange, recomputed analytically)
  // d_in[2] = attention_mask (causal, recomputed analytically)
  const float* Wq = (const float*)d_in[3];
  const float* Wk = (const float*)d_in[4];
  const float* Wv = (const float*)d_in[5];

  char* p = (char*)d_ws;
  __bf16* Xb = (__bf16*)p;  p += (size_t)ROWS * HH * 2;          // 32 MB
  __bf16* Wc = (__bf16*)p;  p += (size_t)HH * NCAT * 2;          // 48 MB
  __bf16* T  = (__bf16*)p;  p += (size_t)ROWS * NCAT * 2;        // 48 MB
  __bf16* Qb = (__bf16*)p;  p += (size_t)BB * NH  * SS * HD * 2; // 32 MB
  __bf16* Kb = (__bf16*)p;  p += (size_t)BB * NKV * SS * HD * 2; //  8 MB
  __bf16* Vb = (__bf16*)p;                                       //  8 MB

  cvt_x<<<(ROWS * HH) / 256, 256, 0, stream>>>(X, Xb);
  cvt_w<<<(HH * NCAT) / 256, 256, 0, stream>>>(Wq, Wk, Wv, Wc);
  qkv_gemm<<<dim3(ROWS / 128, NCAT / 128), 256, 0, stream>>>(Xb, Wc, T);
  rope_pack<<<(ROWS * 48 * 64) / 256, 256, 0, stream>>>(T, Qb, Kb, Vb);
  attn<<<dim3(SS / 128, NH, BB), 256, 0, stream>>>(Qb, Kb, Vb, (float*)d_out);
}
